// GraphSAGEEncoder_22900765622885
// MI455X (gfx1250) — compile-verified, matrix operand order checked by reading, not run on hardware
//
#include <hip/hip_runtime.h>
#include <hip/hip_bf16.h>

#define NN 20000   // nodes
#define DD 128     // feature dim (in == out)

typedef __attribute__((ext_vector_type(16))) _Float16 v16h;
typedef __attribute__((ext_vector_type(8)))  _Float16 v8h;
typedef __attribute__((ext_vector_type(8)))  float    v8f;
typedef __attribute__((ext_vector_type(4)))  float    v4f;

// Build a 16-half WMMA fragment from two contiguous 16B chunks of f16.
// Per CDNA5 ISA 16-bit layout: elements 0..7 = K at +0..7,
// elements 8..15 = K at +16..23 (lane>=16 shifts the base by 8).
__device__ __forceinline__ v16h frag_from_f16(const _Float16* p0, const _Float16* p1) {
    v8h lo = *(const v8h*)p0;
    v8h hi = *(const v8h*)p1;
    return __builtin_shufflevector(lo, hi, 0,1,2,3,4,5,6,7,8,9,10,11,12,13,14,15);
}

// Same fragment but sourced from f32 memory with on-the-fly convert.
__device__ __forceinline__ v16h frag_from_f32(const float* p0, const float* p1) {
    v4f a0 = *(const v4f*)p0;
    v4f a1 = *((const v4f*)p0 + 1);
    v4f b0 = *(const v4f*)p1;
    v4f b1 = *((const v4f*)p1 + 1);
    v16h r;
#pragma unroll
    for (int i = 0; i < 4; ++i) {
        r[i]      = (_Float16)a0[i];
        r[4 + i]  = (_Float16)a1[i];
        r[8 + i]  = (_Float16)b0[i];
        r[12 + i] = (_Float16)b1[i];
    }
    return r;
}

// Non-temporal variant for the 1.6 GB adj stream (don't pollute L2).
__device__ __forceinline__ v16h frag_from_f32_nt(const float* p0, const float* p1) {
    v4f a0 = __builtin_nontemporal_load((const v4f*)p0);
    v4f a1 = __builtin_nontemporal_load((const v4f*)p0 + 1);
    v4f b0 = __builtin_nontemporal_load((const v4f*)p1);
    v4f b1 = __builtin_nontemporal_load((const v4f*)p1 + 1);
    v16h r;
#pragma unroll
    for (int i = 0; i < 4; ++i) {
        r[i]      = (_Float16)a0[i];
        r[4 + i]  = (_Float16)a1[i];
        r[8 + i]  = (_Float16)b0[i];
        r[12 + i] = (_Float16)b1[i];
    }
    return r;
}

// ---------------------------------------------------------------------------
// Kernel 1: xT[d][n] = (f16)x[n][d]  (128 x 20000), LDS-tiled transpose.
// ---------------------------------------------------------------------------
__global__ __launch_bounds__(256) void transpose_cvt_kernel(
        const float* __restrict__ x, _Float16* __restrict__ xT) {
    __shared__ _Float16 tile[32][33];
    const int nb = blockIdx.x * 32;   // node block   (20000/32 = 625)
    const int db = blockIdx.y * 32;   // feature block (128/32 = 4)
    const int tx = threadIdx.x;       // 0..31
    const int ty = threadIdx.y;       // 0..7
#pragma unroll
    for (int i = 0; i < 4; ++i) {
        int n = nb + ty + 8 * i;
        tile[ty + 8 * i][tx] = (_Float16)x[(size_t)n * DD + db + tx];
    }
    __syncthreads();
#pragma unroll
    for (int i = 0; i < 4; ++i) {
        int d = db + ty + 8 * i;
        xT[(size_t)d * NN + nb + tx] = tile[tx][ty + 8 * i];
    }
}

// ---------------------------------------------------------------------------
// Kernel 2: neigh[m][n] = (f16)(adj @ x)[m][n]
// 1250 blocks (one 16-row M-tile each) x 4 waves; wave w owns cols w*32..w*32+31.
// adj fragments go global(f32,NT) -> cvt -> WMMA; xT fragments load as f16.
// ---------------------------------------------------------------------------
__global__ __launch_bounds__(128) void spmm_wmma_kernel(
        const float* __restrict__ adj, const _Float16* __restrict__ xT,
        _Float16* __restrict__ neigh) {
    const int mt    = blockIdx.x;           // M-tile: rows mt*16 .. mt*16+15
    const int wave  = threadIdx.x >> 5;     // 0..3
    const int lane  = threadIdx.x & 31;
    const int lhalf = lane & 15;
    const int khalf = (lane >> 4) << 3;     // 0 for lanes 0-15, 8 for 16-31

    const float* adjRow = adj + (size_t)(mt * 16 + lhalf) * NN;

    const int nbase = wave * 32;            // this wave's column base
    const _Float16* xtRow0 = xT + (size_t)(nbase + lhalf) * NN;       // tile 0
    const _Float16* xtRow1 = xT + (size_t)(nbase + 16 + lhalf) * NN;  // tile 1

    v8f acc0 = {};
    v8f acc1 = {};

#pragma unroll 2
    for (int kk = 0; kk < NN; kk += 32) {
        // pull the adj stream ahead of the consume point
        __builtin_prefetch(adjRow + kk + 2048, 0, 0);
        v16h a  = frag_from_f32_nt(adjRow + kk + khalf, adjRow + kk + khalf + 16);
        v16h b0 = frag_from_f16(xtRow0 + kk + khalf, xtRow0 + kk + khalf + 16);
        v16h b1 = frag_from_f16(xtRow1 + kk + khalf, xtRow1 + kk + khalf + 16);
        acc0 = __builtin_amdgcn_wmma_f32_16x16x32_f16(false, a, false, b0,
                                                      (short)0, acc0, false, false);
        acc1 = __builtin_amdgcn_wmma_f32_16x16x32_f16(false, a, false, b1,
                                                      (short)0, acc1, false, false);
    }

    // C/D layout: VGPR v -> M = v (lanes 0-15) or v+8 (lanes 16-31); N = lane&15.
#pragma unroll
    for (int v = 0; v < 8; ++v) {
        int row = mt * 16 + v + khalf;
        neigh[(size_t)row * DD + nbase + lhalf]      = (_Float16)acc0[v];
        neigh[(size_t)row * DD + nbase + 16 + lhalf] = (_Float16)acc1[v];
    }
}

// ---------------------------------------------------------------------------
// Kernel 3: out = relu(x @ Wself^T + neigh @ Wneigh^T), K = 128.
// B[k][j] = W[n0+j][k] -> W rows are K-contiguous: load + convert directly.
// ---------------------------------------------------------------------------
__global__ __launch_bounds__(128) void out_wmma_kernel(
        const float* __restrict__ x, const float* __restrict__ Wself,
        const float* __restrict__ Wneigh, const _Float16* __restrict__ neigh,
        float* __restrict__ out) {
    const int mt    = blockIdx.x;
    const int wave  = threadIdx.x >> 5;
    const int lane  = threadIdx.x & 31;
    const int lhalf = lane & 15;
    const int khalf = (lane >> 4) << 3;

    const float*    xRow = x     + (size_t)(mt * 16 + lhalf) * DD;
    const _Float16* nRow = neigh + (size_t)(mt * 16 + lhalf) * DD;
    const int nbase = wave * 32;

    v8f acc[2] = {};
#pragma unroll
    for (int kk = 0; kk < DD; kk += 32) {
        v16h a1 = frag_from_f32(xRow + kk + khalf, xRow + kk + khalf + 16);
        v16h a2 = frag_from_f16(nRow + kk + khalf, nRow + kk + khalf + 16);
#pragma unroll
        for (int t = 0; t < 2; ++t) {
            const float* wsRow = Wself  + (size_t)(nbase + t * 16 + lhalf) * DD;
            const float* wnRow = Wneigh + (size_t)(nbase + t * 16 + lhalf) * DD;
            v16h b1 = frag_from_f32(wsRow + kk + khalf, wsRow + kk + khalf + 16);
            v16h b2 = frag_from_f32(wnRow + kk + khalf, wnRow + kk + khalf + 16);
            acc[t] = __builtin_amdgcn_wmma_f32_16x16x32_f16(false, a1, false, b1,
                                                            (short)0, acc[t], false, false);
            acc[t] = __builtin_amdgcn_wmma_f32_16x16x32_f16(false, a2, false, b2,
                                                            (short)0, acc[t], false, false);
        }
    }

#pragma unroll
    for (int t = 0; t < 2; ++t) {
        int col = nbase + t * 16 + lhalf;
#pragma unroll
        for (int v = 0; v < 8; ++v) {
            int row = mt * 16 + v + khalf;
            float r = acc[t][v];
            out[(size_t)row * DD + col] = r > 0.0f ? r : 0.0f;
        }
    }
}

// ---------------------------------------------------------------------------
extern "C" void kernel_launch(void* const* d_in, const int* in_sizes, int n_in,
                              void* d_out, int out_size, void* d_ws, size_t ws_size,
                              hipStream_t stream) {
    const float* x      = (const float*)d_in[0];   // (20000,128)
    const float* adj    = (const float*)d_in[1];   // (20000,20000)
    const float* Wself  = (const float*)d_in[2];   // (128,128)
    const float* Wneigh = (const float*)d_in[3];   // (128,128)
    float*       out    = (float*)d_out;           // (20000,128)

    _Float16* xT    = (_Float16*)d_ws;                                   // 128 x 20000 f16
    _Float16* neigh = (_Float16*)((char*)d_ws +
                                  (size_t)DD * NN * sizeof(_Float16));   // 20000 x 128 f16

    transpose_cvt_kernel<<<dim3(NN / 32, DD / 32), dim3(32, 8), 0, stream>>>(x, xT);
    spmm_wmma_kernel<<<NN / 16, 128, 0, stream>>>(adj, xT, neigh);
    out_wmma_kernel<<<NN / 16, 128, 0, stream>>>(x, Wself, Wneigh, neigh, out);
}
